// UncertaintySurrogateModel_81209241632904
// MI455X (gfx1250) — matrix operand order repaired and checked by reading, not compiled, not run on hardware
//
#include <hip/hip_runtime.h>

// ---------------------------------------------------------------------------
// MI455X (gfx1250) implementation. wave32, WMMA f32_16x16x32_f16 everywhere,
// async global->LDS DMA for fp16 tile staging, 2x2 register tiling per wave.
// ---------------------------------------------------------------------------

typedef _Float16 half_t;
typedef __attribute__((ext_vector_type(16))) _Float16 v16h;
typedef __attribute__((ext_vector_type(8)))  _Float16 v8h;
typedef __attribute__((ext_vector_type(8)))  float    v8f;

#define DEV __device__ __forceinline__

DEV v8f wmma_f16(v16h a, v16h b, v8f c) {
  // D = A(16x32 f16) x B(32x16 f16) + C(16x16 f32)
  return __builtin_amdgcn_wmma_f32_16x16x32_f16(false, a, false, b, (short)0, c,
                                                false, false);
}

DEV v16h ld16(const half_t* p0, const half_t* p1) {
  union { v16h v; v8h h[2]; } u;
  u.h[0] = *(const v8h*)p0;
  u.h[1] = *(const v8h*)p1;
  return u.v;
}

// fp32 -> fp16 cooperative staging (conversion required, plain path)
DEV void stage_input(const float* __restrict__ ip, half_t* s, int n, int tid) {
  for (int i = tid; i < n; i += 256) s[i] = (half_t)ip[i];
}

// fp16 -> fp16 staging via asynchronous global->LDS DMA (ASYNCcnt tracked).
// 16 bytes (8 halves) per lane; LDS address = low 32 bits of shared pointer.
DEV void stage_input(const half_t* __restrict__ ip, half_t* s, int n, int tid) {
  int chunks = n >> 3;
  for (int i = tid; i < chunks; i += 256) {
    unsigned loff = (unsigned)(size_t)(s + i * 8);
    const void* g = (const void*)(ip + i * 8);
    asm volatile("global_load_async_to_lds_b128 %0, %1, off"
                 :: "v"(loff), "v"(g) : "memory");
  }
  asm volatile("s_wait_asynccnt 0" ::: "memory");
}

// ---------------------------------------------------------------------------
// Generic WMMA GEMM:  Y[M,N] (=|+=) act( A[M,K] @ W[N,K]^T + bias + addend )
// block tile 64x64, 128 threads = 4 waves, each wave owns a 32x32 C tile
// (2x2 register tiling: 4 WMMA per 8 LDS B128 fragment loads).
// flags: 1 = relu, 2 = accumulate into Y
// ---------------------------------------------------------------------------
__global__ __launch_bounds__(128) void gemm_wmma(
    const float* __restrict__ A, int lda,
    const float* __restrict__ W, int ldw,
    const float* __restrict__ bias,
    const float* __restrict__ addend, int ldadd,
    float* __restrict__ Y, int ldy,
    int M, int N, int Kd, int flags)
{
  __shared__ __attribute__((aligned(16))) half_t As[64][72];
  __shared__ __attribute__((aligned(16))) half_t Ws[64][72];
  const int bm = blockIdx.x * 64;
  const int bn = blockIdx.y * 64;
  const int tid = threadIdx.x;
  const int lane = tid & 31;
  const int wave = tid >> 5;
  const int wm = wave & 1;   // 32-row subtile
  const int wn = wave >> 1;  // 32-col subtile

  v8f acc00 = {0.f,0.f,0.f,0.f,0.f,0.f,0.f,0.f};
  v8f acc01 = {0.f,0.f,0.f,0.f,0.f,0.f,0.f,0.f};
  v8f acc10 = {0.f,0.f,0.f,0.f,0.f,0.f,0.f,0.f};
  v8f acc11 = {0.f,0.f,0.f,0.f,0.f,0.f,0.f,0.f};

  for (int k0 = 0; k0 < Kd; k0 += 64) {
    // cooperative stage (fp32 -> fp16), branchless: clamped load + select
    for (int i = tid; i < 64 * 64; i += 128) {
      int rr = i >> 6, cc = i & 63;
      int gr = bm + rr, gc = k0 + cc;
      int gcc = gc < Kd ? gc : Kd - 1;
      bool kok = gc < Kd;
      int ga = gr < M ? gr : M - 1;
      float va = A[(size_t)ga * lda + gcc];
      As[rr][cc] = (half_t)((gr < M && kok) ? va : 0.f);
      int wr = bn + rr;
      int wc = wr < N ? wr : N - 1;
      float vw = W[(size_t)wc * ldw + gcc];
      Ws[rr][cc] = (half_t)((wr < N && kok) ? vw : 0.f);
    }
    // prefetch next K panel of A into cache
    if (k0 + 64 < Kd) {
      int pr = bm + (tid & 63);
      if (pr < M) __builtin_prefetch(&A[(size_t)pr * lda + k0 + 64], 0, 1);
    }
    __syncthreads();

    #pragma unroll
    for (int kk = 0; kk < 64; kk += 32) {
      // A frags: lanes<16 hold K {kk..kk+7, kk+16..kk+23}; lanes>=16 shift by 8
      int ar = wm * 32 + (lane & 15);
      int ab = kk + ((lane >> 4) << 3);
      v16h a0 = ld16(&As[ar][ab], &As[ar][ab + 16]);
      v16h a1 = ld16(&As[ar + 16][ab], &As[ar + 16][ab + 16]);
      // B frags: same K layout as A frags
      int br = wn * 32 + (lane & 15);
      int bb = kk + ((lane >> 4) << 3);
      v16h b0 = ld16(&Ws[br][bb], &Ws[br][bb + 16]);
      v16h b1 = ld16(&Ws[br + 16][bb], &Ws[br + 16][bb + 16]);
      acc00 = wmma_f16(a0, b0, acc00);
      acc01 = wmma_f16(a0, b1, acc01);
      acc10 = wmma_f16(a1, b0, acc10);
      acc11 = wmma_f16(a1, b1, acc11);
    }
    __syncthreads();
  }

  // epilogue: C layout -> VGPR r: lane<16 M=r, lane>=16 M=r+8; N = lane&15
  #pragma unroll
  for (int mi = 0; mi < 2; ++mi) {
    #pragma unroll
    for (int ni = 0; ni < 2; ++ni) {
      v8f acc = mi ? (ni ? acc11 : acc10) : (ni ? acc01 : acc00);
      int col = bn + wn * 32 + ni * 16 + (lane & 15);
      int rb  = bm + wm * 32 + mi * 16 + ((lane >> 4) << 3);
      if (col < N) {
        float bv = bias ? bias[col] : 0.f;
        #pragma unroll
        for (int r = 0; r < 8; ++r) {
          int row = rb + r;
          if (row < M) {
            float v = acc[r] + bv;
            if (addend) v += addend[(size_t)row * ldadd + col];
            if (flags & 1) v = fmaxf(v, 0.f);
            if (flags & 2) Y[(size_t)row * ldy + col] += v;
            else           Y[(size_t)row * ldy + col]  = v;
          }
        }
      }
    }
  }
}

// ---------------------------------------------------------------------------
// Implicit-GEMM conv + bias (+ batch stats | + BN + relu + 4x2 maxpool)
// one image per workgroup; M = IH*IW pixels, N = 32 channels, K = CIN*7*3
// mode 0: accumulate per-channel sum/sumsq (LDS ds_add_f32 + global atomics)
// mode 1: normalize with mean/rstd, relu, pool, write pooled output
// fp16 input (conv2) is staged into LDS with async global->LDS DMA.
// ---------------------------------------------------------------------------
template <typename TIN, int CIN, int IH, int IW, int KPAD>
__global__ __launch_bounds__(256) void conv_bn_pool(
    const TIN* __restrict__ input,         // [img][CIN][IH][IW]
    const half_t* __restrict__ wh,         // [32][KPAD] fp16 weights
    const float* __restrict__ cbias,
    float* __restrict__ stats,             // [64] sums (mode 0)
    const float* __restrict__ mean, const float* __restrict__ rstd,
    const float* __restrict__ gam, const float* __restrict__ bet,
    half_t* __restrict__ outh,             // pooled fp16 (conv1) or null
    float*  __restrict__ outf,             // pooled fp32 (conv2) or null
    int mode)
{
  constexpr int NPIX = IH * IW;
  constexpr int MT = NPIX / 16;
  constexpr int NTILES = MT * 2;
  constexpr int PH = IH / 4, PW = IW / 2, POOLN = PH * PW;

  __shared__ __attribute__((aligned(16))) half_t s_in[CIN * IH * IW];
  __shared__ __attribute__((aligned(16))) half_t s_out[32 * NPIX];
  __shared__ int    s_ktab[KPAD];
  __shared__ float  s_stat[64];

  const int img = blockIdx.x;
  const int tid = threadIdx.x;
  const TIN* ip = input + (size_t)img * CIN * IH * IW;

  stage_input(ip, s_in, CIN * IH * IW, tid);
  for (int i = tid; i < KPAD; i += 256) {
    int ic = i / 21, r = i - ic * 21;
    s_ktab[i] = (ic << 8) | ((r / 3) << 4) | (r % 3);
  }
  if (mode == 0) for (int i = tid; i < 64; i += 256) s_stat[i] = 0.f;
  __syncthreads();

  const int lane = tid & 31;
  const int wave = tid >> 5;

  for (int tile = wave; tile < NTILES; tile += 8) {
    int mt = tile >> 1, nt = tile & 1;
    int m0 = mt * 16;
    int p = m0 + (lane & 15);
    int oy = p / IW, ox = p % IW;
    v8f acc = {0.f,0.f,0.f,0.f,0.f,0.f,0.f,0.f};

    for (int k0 = 0; k0 < KPAD; k0 += 32) {
      // A fragment: im2col gather from LDS
      union { v16h v; half_t e[16]; } ua;
      int kb = k0 + ((lane >> 4) << 3);
      #pragma unroll
      for (int j = 0; j < 16; ++j) {
        int k = kb + (j & 7) + ((j >> 3) << 4);
        int e = s_ktab[k];
        int ic = e >> 8, ky = (e >> 4) & 15, kx = e & 15;
        int iy = oy + ky - 3, ix = ox + kx - 1;
        bool ok = (ic < CIN) & (iy >= 0) & (iy < IH) & (ix >= 0) & (ix < IW);
        ua.e[j] = ok ? s_in[ic * IH * IW + iy * IW + ix] : (half_t)0.f;
      }
      // B fragment: K {kb..kb+7, kb+16..kb+23} of one output channel
      int ch = nt * 16 + (lane & 15);
      int bb = k0 + ((lane >> 4) << 3);
      const half_t* wp = &wh[(size_t)ch * KPAD + bb];
      v16h bf = ld16(wp, wp + 16);
      acc = wmma_f16(ua.v, bf, acc);
    }

    int ch = nt * 16 + (lane & 15);
    float bv = cbias[ch];
    int prow = m0 + ((lane >> 4) << 3);
    if (mode == 0) {
      float s = 0.f, s2 = 0.f;
      #pragma unroll
      for (int r = 0; r < 8; ++r) { float v = acc[r] + bv; s += v; s2 += v * v; }
      atomicAdd(&s_stat[ch], s);         // ds_add_f32
      atomicAdd(&s_stat[32 + ch], s2);
    } else {
      float mch = mean[ch], rch = rstd[ch], gch = gam[ch], bch = bet[ch];
      #pragma unroll
      for (int r = 0; r < 8; ++r) {
        float v = acc[r] + bv;
        v = (v - mch) * rch * gch + bch;
        s_out[ch * NPIX + prow + r] = (half_t)fmaxf(v, 0.f);
      }
    }
  }
  __syncthreads();

  if (mode == 0) {
    for (int i = tid; i < 64; i += 256) atomicAdd(&stats[i], s_stat[i]);
  } else {
    for (int i = tid; i < 32 * POOLN; i += 256) {
      int ch = i / POOLN, pp = i % POOLN;
      int py = pp / PW, px = pp % PW;
      float mx = -1e30f;
      #pragma unroll
      for (int dy = 0; dy < 4; ++dy)
        #pragma unroll
        for (int dx = 0; dx < 2; ++dx)
          mx = fmaxf(mx, (float)s_out[ch * NPIX + (py * 4 + dy) * IW + px * 2 + dx]);
      if (outh) outh[(size_t)img * 32 * POOLN + i] = (half_t)mx;
      else      outf[(size_t)img * 32 * POOLN + i] = mx;
    }
  }
}

// ---------------------------------------------------------------------------
// small helper kernels
// ---------------------------------------------------------------------------
__global__ void cvt_w(const float* __restrict__ src, half_t* __restrict__ dst,
                      int oc, int ksrc, int kdst) {
  int i = blockIdx.x * 256 + threadIdx.x;
  if (i >= oc * kdst) return;
  int o = i / kdst, k = i % kdst;
  dst[i] = (k < ksrc) ? (half_t)src[o * ksrc + k] : (half_t)0.f;
}

__global__ void finalize_stats(const float* __restrict__ stats,
                               float* __restrict__ mean, float* __restrict__ rstd,
                               float count) {
  int c = threadIdx.x;
  if (c < 32) {
    float m = stats[c] / count;
    float v = stats[32 + c] / count - m * m;
    mean[c] = m;
    rstd[c] = rsqrtf(v + 1e-5f);
  }
}

// (B*L, S, H) -> (S, B*L, H)
__global__ void reorder_SBH(const float* __restrict__ in, float* __restrict__ out) {
  int idx = blockIdx.x * 256 + threadIdx.x;
  if (idx >= 4096 * 256) return;
  int h = idx & 255, r = idx >> 8;
  int s = r & 15, bl = r >> 4;
  out[((s * 256) + bl) * 256 + h] = in[idx];
}

// (B, L, H) -> (L, B, H)
__global__ void reorder_useq(const float* __restrict__ in, float* __restrict__ out) {
  int idx = blockIdx.x * 256 + threadIdx.x;
  if (idx >= 256 * 256) return;
  int h = idx & 255, r = idx >> 8;
  int b = r >> 3, l = r & 7;
  out[((l * 32 + b) << 8) + h] = in[idx];
}

// tc (B=32, 24, 13) -> enc rows (b*8+l) x13 ; dec rows (t*32+b) x13
__global__ void slice_tc(const float* __restrict__ tc,
                         float* __restrict__ enc, float* __restrict__ dec) {
  int i = blockIdx.x * 256 + threadIdx.x;
  if (i < 256 * 13) {
    int r = i / 13, c = i % 13;
    int b = r / 8, l = r % 8;
    enc[i] = tc[(b * 24 + l) * 13 + c];
  }
  if (i < 512 * 13) {
    int r = i / 13, c = i % 13;
    int t = r / 32, b = r % 32;
    dec[i] = tc[(b * 24 + 8 + t) * 13 + c];
  }
}

__global__ void copy_mat(const float* __restrict__ src, int lds,
                         float* __restrict__ dst, int ldd, int rows, int cols) {
  int i = blockIdx.x * 256 + threadIdx.x;
  if (i >= rows * cols) return;
  int r = i / cols, c = i % cols;
  dst[(size_t)r * ldd + c] = src[(size_t)r * lds + c];
}

// fused LSTM gate nonlinearity + state update (PyTorch gate order i,f,g,o)
__global__ void lstm_update(const float* __restrict__ g, int ldg,
                            float* __restrict__ h, float* __restrict__ c,
                            int M, int H,
                            float* __restrict__ yout, int ldy,
                            const float* __restrict__ wlast, int wstride,
                            const float* __restrict__ fb) {
  int idx = blockIdx.x * 256 + threadIdx.x;
  if (idx >= M * H) return;
  int m = idx / H, j = idx % H;
  float ai = 0.f, af = 0.f, ag = 0.f, ao = 0.f;
  if (fb) {
    float o = fb[m];
    ai = o * wlast[(size_t)j * wstride];
    af = o * wlast[(size_t)(H + j) * wstride];
    ag = o * wlast[(size_t)(2 * H + j) * wstride];
    ao = o * wlast[(size_t)(3 * H + j) * wstride];
  }
  const float* gr = g + (size_t)m * ldg;
  float gi = gr[j] + ai, gf = gr[H + j] + af;
  float gg = gr[2 * H + j] + ag, go = gr[3 * H + j] + ao;
  float si = 1.f / (1.f + __expf(-gi));
  float sf = 1.f / (1.f + __expf(-gf));
  float so = 1.f / (1.f + __expf(-go));
  float cc = sf * c[idx] + si * tanhf(gg);
  float hh = so * tanhf(cc);
  c[idx] = cc;
  h[idx] = hh;
  if (yout) yout[(size_t)m * ldy + j] = hh;
}

// softmax(pi), exp(sigma), mdn mean feedback, transposed store to d_out
__global__ void mdn_finalize(const float* __restrict__ pi,
                             const float* __restrict__ mu,
                             const float* __restrict__ sg,
                             float* __restrict__ out, float* __restrict__ fb,
                             int t) {
  int b = threadIdx.x;
  if (b >= 32) return;
  float p[5];
  float mx = -1e30f;
  #pragma unroll
  for (int k = 0; k < 5; ++k) { p[k] = pi[b * 5 + k]; mx = fmaxf(mx, p[k]); }
  float sum = 0.f;
  #pragma unroll
  for (int k = 0; k < 5; ++k) { p[k] = __expf(p[k] - mx); sum += p[k]; }
  float inv = 1.f / sum, acc = 0.f;
  float* cO = out, *mO = out + 2560, *sO = out + 5120;
  #pragma unroll
  for (int k = 0; k < 5; ++k) {
    float c = p[k] * inv;
    float mm = mu[b * 5 + k];
    cO[(b * 16 + t) * 5 + k] = c;
    mO[(b * 16 + t) * 5 + k] = mm;
    sO[(b * 16 + t) * 5 + k] = __expf(sg[b * 5 + k]);
    acc += c * mm;
  }
  fb[b] = acc;
}

// ---------------------------------------------------------------------------
// host side
// ---------------------------------------------------------------------------
struct LRBp  { const float *w1, *b1, *w2, *b2, *ws, *bs; };
struct LSTMp { const float *wih, *whh, *b; };

static void launch_gemm(hipStream_t s, const float* A, int lda, const float* W,
                        int ldw, const float* bias, const float* add, int ldadd,
                        float* Y, int ldy, int M, int N, int K, int flags) {
  dim3 g((unsigned)((M + 63) / 64), (unsigned)((N + 63) / 64));
  gemm_wmma<<<g, 128, 0, s>>>(A, lda, W, ldw, bias, add, ldadd, Y, ldy, M, N, K,
                              flags);
}

// LinearResBlock: out = fc2(relu(fc1 x)) + skip(x)
static void lrb_run(hipStream_t s, const float* X, int M, int K, const LRBp& p,
                    float* tmp, float* out, int Hh, int O) {
  launch_gemm(s, X, K, p.w1, K, p.b1, nullptr, 0, tmp, Hh, M, Hh, K, 1);
  launch_gemm(s, X, K, p.ws, K, p.bs, nullptr, 0, out, O, M, O, K, 0);
  launch_gemm(s, tmp, Hh, p.w2, Hh, p.b2, nullptr, 0, out, O, M, O, Hh, 2);
}

extern "C" void kernel_launch(void* const* d_in, const int* in_sizes, int n_in,
                              void* d_out, int out_size, void* d_ws,
                              size_t ws_size, hipStream_t stream) {
  (void)in_sizes; (void)n_in; (void)out_size; (void)ws_size;
  const float* obs = (const float*)d_in[0];
  const float* tc  = (const float*)d_in[1];
  int pc = 2;
  auto F = [&]() { return (const float*)d_in[pc++]; };
  const float* c1w = F(); const float* c1b = F();
  const float* bn1g = F(); const float* bn1b = F();
  const float* c2w = F(); const float* c2b = F();
  const float* bn2g = F(); const float* bn2b = F();
  auto LRBP  = [&]() { LRBp p; p.w1=F(); p.b1=F(); p.w2=F(); p.b2=F(); p.ws=F(); p.bs=F(); return p; };
  auto LSTMP = [&]() { LSTMp p; p.wih=F(); p.whh=F(); p.b=F(); return p; };
  LRBp  l_in = LRBP();
  LSTMp l0f = LSTMP(), l0b = LSTMP(), l1f = LSTMP(), l1b = LSTMP();
  LRBp  lo = LRBP(), tco = LRBP(), ui = LRBP();
  LSTMp ul = LSTMP();
  LRBp  uo = LRBP(), di = LRBP();
  LSTMp dc = LSTMP();
  LRBp  dob = LRBP();
  const float* mfc_w = F(); const float* mfc_b = F();
  const float* mpi_w = F(); const float* mpi_b = F();
  const float* mmu_w = F(); const float* mmu_b = F();
  const float* msg_w = F(); const float* msg_b = F();

  // ---- workspace bump allocator ----
  char* base = (char*)d_ws;
  size_t off = 0;
  auto allocf = [&](size_t n) { float*  p = (float*) (base + off); off += (n * 4 + 255) & ~(size_t)255; return p; };
  auto alloch = [&](size_t n) { half_t* p = (half_t*)(base + off); off += (n * 2 + 255) & ~(size_t)255; return p; };

  float* stats1 = allocf(64); float* mean1 = allocf(32); float* rstd1 = allocf(32);
  float* stats2 = allocf(64); float* mean2 = allocf(32); float* rstd2 = allocf(32);
  half_t* w1h  = alloch(32 * 432);
  half_t* w2h  = alloch(32 * 672);
  half_t* pool1 = alloch((size_t)4096 * 1536);
  float* feat  = allocf((size_t)4096 * 192);
  float* tmpA  = allocf((size_t)4096 * 256);
  float* linb  = allocf((size_t)4096 * 256);
  float* xseq  = allocf((size_t)4096 * 256);
  float* Xw0f  = allocf((size_t)4096 * 1024);
  float* Xw0b  = allocf((size_t)4096 * 1024);
  float* y01   = allocf((size_t)16 * 256 * 512);
  float* Xw1f  = allocf((size_t)4096 * 1024);
  float* Xw1b  = allocf((size_t)4096 * 1024);
  float* gates = allocf((size_t)256 * 1024);
  float* hst = allocf(256 * 256); float* cst = allocf(256 * 256);
  float* lower = allocf(256 * 512);
  float* lout  = allocf(256 * 256);
  float* tcbin = allocf(256 * 13);
  float* tcb   = allocf(256 * 256);
  float* ucat  = allocf(256 * 512);
  float* uout  = allocf(256 * 256);
  float* useq  = allocf(256 * 256);
  float* Xwu   = allocf(256 * 1024);
  float* uh = allocf(32 * 256); float* uc = allocf(32 * 256);
  float* upper = allocf(32 * 256);
  float* tcd   = allocf(512 * 13);
  float* tdec  = allocf(512 * 256);
  float* Xwd   = allocf((size_t)512 * 1024);
  float* dh = allocf(32 * 256); float* dcc = allocf(32 * 256);
  float* dfb = allocf(32);
  float* dgates = allocf(32 * 1024);
  float* dtmp = allocf(32 * 256);
  float* dd   = allocf(32 * 256);
  float* dhm  = allocf(32 * 256);
  float* dpi = allocf(32 * 5); float* dmu = allocf(32 * 5); float* dsg = allocf(32 * 5);

  // ---- conv stack: stats pass, then normalize+pool pass (recompute) ----
  cvt_w<<<(32 * 432 + 255) / 256, 256, 0, stream>>>(c1w, w1h, 32, 420, 432);
  cvt_w<<<(32 * 672 + 255) / 256, 256, 0, stream>>>(c2w, w2h, 32, 672, 672);

  hipMemsetAsync(stats1, 0, 64 * 4, stream);
  conv_bn_pool<float, 20, 32, 12, 432><<<4096, 256, 0, stream>>>(
      obs, w1h, c1b, stats1, nullptr, nullptr, nullptr, nullptr, nullptr,
      nullptr, 0);
  finalize_stats<<<1, 32, 0, stream>>>(stats1, mean1, rstd1, 4096.f * 384.f);
  conv_bn_pool<float, 20, 32, 12, 432><<<4096, 256, 0, stream>>>(
      obs, w1h, c1b, nullptr, mean1, rstd1, bn1g, bn1b, pool1, nullptr, 1);

  hipMemsetAsync(stats2, 0, 64 * 4, stream);
  conv_bn_pool<half_t, 32, 8, 6, 672><<<4096, 256, 0, stream>>>(
      pool1, w2h, c2b, stats2, nullptr, nullptr, nullptr, nullptr, nullptr,
      nullptr, 0);
  finalize_stats<<<1, 32, 0, stream>>>(stats2, mean2, rstd2, 4096.f * 48.f);
  conv_bn_pool<half_t, 32, 8, 6, 672><<<4096, 256, 0, stream>>>(
      pool1, w2h, c2b, nullptr, mean2, rstd2, bn2g, bn2b, nullptr, feat, 1);

  // ---- l_in LRB + reorder to (S, B*L, H) ----
  lrb_run(stream, feat, 4096, 192, l_in, tmpA, linb, 256, 256);
  reorder_SBH<<<4096, 256, 0, stream>>>(linb, xseq);

  // ---- BiLSTM layer 0: batch the x@Wih for all 16 steps ----
  launch_gemm(stream, xseq, 256, l0f.wih, 256, nullptr, nullptr, 0, Xw0f, 1024,
              4096, 1024, 256, 0);
  launch_gemm(stream, xseq, 256, l0b.wih, 256, nullptr, nullptr, 0, Xw0b, 1024,
              4096, 1024, 256, 0);

  auto run_lstm = [&](const float* Xw, const LSTMp& lp, int M, int T, bool rev,
                      float* ybase, int ycols, int ycoloff) {
    hipMemsetAsync(hst, 0, (size_t)M * 256 * 4, stream);
    hipMemsetAsync(cst, 0, (size_t)M * 256 * 4, stream);
    for (int s = 0; s < T; ++s) {
      int t = rev ? (T - 1 - s) : s;
      launch_gemm(stream, hst, 256, lp.whh, 256, lp.b,
                  Xw + (size_t)t * M * 1024, 1024, gates, 1024, M, 1024, 256, 0);
      float* yo = ybase ? (ybase + (size_t)t * M * ycols + ycoloff) : nullptr;
      lstm_update<<<(M * 256 + 255) / 256, 256, 0, stream>>>(
          gates, 1024, hst, cst, M, 256, yo, ycols, nullptr, 0, nullptr);
    }
  };

  run_lstm(Xw0f, l0f, 256, 16, false, y01, 512, 0);
  run_lstm(Xw0b, l0b, 256, 16, true,  y01, 512, 256);

  // ---- BiLSTM layer 1 (K=512), only final hidden states needed ----
  launch_gemm(stream, y01, 512, l1f.wih, 512, nullptr, nullptr, 0, Xw1f, 1024,
              4096, 1024, 512, 0);
  launch_gemm(stream, y01, 512, l1b.wih, 512, nullptr, nullptr, 0, Xw1b, 1024,
              4096, 1024, 512, 0);
  run_lstm(Xw1f, l1f, 256, 16, false, nullptr, 0, 0);
  copy_mat<<<(256 * 256 + 255) / 256, 256, 0, stream>>>(hst, 256, lower, 512,
                                                        256, 256);
  run_lstm(Xw1b, l1b, 256, 16, true, nullptr, 0, 0);
  copy_mat<<<(256 * 256 + 255) / 256, 256, 0, stream>>>(hst, 256, lower + 256,
                                                        512, 256, 256);

  // ---- upper encoding ----
  lrb_run(stream, lower, 256, 512, lo, tmpA, lout, 256, 256);
  slice_tc<<<(512 * 13 + 255) / 256, 256, 0, stream>>>(tc, tcbin, tcd);
  lrb_run(stream, tcbin, 256, 13, tco, tmpA, tcb, 256, 256);
  copy_mat<<<(256 * 256 + 255) / 256, 256, 0, stream>>>(tcb, 256, ucat, 512,
                                                        256, 256);
  copy_mat<<<(256 * 256 + 255) / 256, 256, 0, stream>>>(lout, 256, ucat + 256,
                                                        512, 256, 256);
  lrb_run(stream, ucat, 256, 512, ui, tmpA, uout, 256, 256);
  reorder_useq<<<256, 256, 0, stream>>>(uout, useq);

  launch_gemm(stream, useq, 256, ul.wih, 256, nullptr, nullptr, 0, Xwu, 1024,
              256, 1024, 256, 0);
  hipMemsetAsync(uh, 0, 32 * 256 * 4, stream);
  hipMemsetAsync(uc, 0, 32 * 256 * 4, stream);
  for (int t = 0; t < 8; ++t) {
    launch_gemm(stream, uh, 256, ul.whh, 256, ul.b, Xwu + (size_t)t * 32 * 1024,
                1024, gates, 1024, 32, 1024, 256, 0);
    lstm_update<<<32, 256, 0, stream>>>(gates, 1024, uh, uc, 32, 256, nullptr,
                                        0, nullptr, 0, nullptr);
  }
  lrb_run(stream, uh, 32, 256, uo, tmpA, upper, 256, 256);

  // ---- decoder: precompute dec_in LRB + x@Wih[:, :256] for all 16 steps ----
  lrb_run(stream, tcd, 512, 13, di, tmpA, tdec, 256, 256);
  launch_gemm(stream, tdec, 256, dc.wih, 257, nullptr, nullptr, 0, Xwd, 1024,
              512, 1024, 256, 0);

  copy_mat<<<(32 * 256 + 255) / 256, 256, 0, stream>>>(upper, 256, dh, 256, 32,
                                                       256);
  hipMemsetAsync(dcc, 0, 32 * 256 * 4, stream);
  hipMemsetAsync(dfb, 0, 32 * 4, stream);

  for (int t = 0; t < 16; ++t) {
    launch_gemm(stream, dh, 256, dc.whh, 256, dc.b, Xwd + (size_t)t * 32 * 1024,
                1024, dgates, 1024, 32, 1024, 256, 0);
    // rank-1 feedback (out * wih[:,256]) folded into gate update
    lstm_update<<<32, 256, 0, stream>>>(dgates, 1024, dh, dcc, 32, 256, nullptr,
                                        0, dc.wih + 256, 257, dfb);
    lrb_run(stream, dh, 32, 256, dob, dtmp, dd, 256, 256);
    launch_gemm(stream, dd, 256, mfc_w, 256, mfc_b, nullptr, 0, dhm, 256, 32,
                256, 256, 1);
    launch_gemm(stream, dhm, 256, mpi_w, 256, mpi_b, nullptr, 0, dpi, 5, 32, 5,
                256, 0);
    launch_gemm(stream, dhm, 256, mmu_w, 256, mmu_b, nullptr, 0, dmu, 5, 32, 5,
                256, 0);
    launch_gemm(stream, dhm, 256, msg_w, 256, msg_b, nullptr, 0, dsg, 5, 32, 5,
                256, 0);
    mdn_finalize<<<1, 32, 0, stream>>>(dpi, dmu, dsg, (float*)d_out, dfb, t);
  }
}
